// ABCNN_44830868635746
// MI455X (gfx1250) — matrix operand order repaired and checked by reading, not compiled, not run on hardware
//
#include <hip/hip_runtime.h>
#include <hip/hip_bf16.h>
#include <math.h>

// ---------------------------------------------------------------------------
// ABCNN forward for MI455X (gfx1250, wave32). All GEMM-shaped work runs on
// v_wmma_f32_16x16x32_f16 (f16 inputs, f32 accumulate), with register-level
// A-fragment reuse: each wave owns one M-tile and 4-5 N-tiles.
// ---------------------------------------------------------------------------

typedef _Float16 half_t;
typedef __attribute__((ext_vector_type(16))) _Float16 v16h;
typedef __attribute__((ext_vector_type(8)))  float    v8f;

union frag16 { v16h h; uint4 u[2]; };   // 32 bytes: one WMMA 16-bit operand

#define DEV static __device__ __forceinline__

DEV v8f wmma32(const frag16& a, const frag16& b, v8f c) {
    // D = A(16x32 f16) * B(32x16 f16) + C(16x16 f32)
    return __builtin_amdgcn_wmma_f32_16x16x32_f16(
        /*neg_a=*/false, a.h, /*neg_b=*/false, b.h,
        /*c_mod=*/(short)0, c, /*reuse_a=*/false, /*reuse_b=*/false);
}

// A-fragment: lane holds row M = lane%16; K chunks at hi*8 and hi*8+16 (halves)
DEV frag16 load_a(const half_t* row, int kbase, int hi) {
    frag16 f;
    f.u[0] = *(const uint4*)(row + kbase + hi * 8);
    f.u[1] = *(const uint4*)(row + kbase + hi * 8 + 16);
    return f;
}
// B-fragment for C = A * B^T: lane holds column N = lane%16 (a row of B^T),
// K chunks at hi*8 and hi*8+16 (same order as A).
DEV frag16 load_b(const half_t* row, int kbase, int hi) {
    frag16 f;
    f.u[0] = *(const uint4*)(row + kbase + hi * 8);
    f.u[1] = *(const uint4*)(row + kbase + hi * 8 + 16);
    return f;
}

static constexpr int   Bb = 128, Ss = 64, D0 = 768, Hh = 256, Kk = 7, Cc = 3;
static constexpr int   Wp = Ss + Kk - 1;      // 70 conv output positions
static constexpr float EPS = 1e-12f;

// ---------------------------------------------------------------------------
// Small elementwise / reduction kernels
// ---------------------------------------------------------------------------

__global__ void k_f32_to_f16(const float* __restrict__ src,
                             half_t* __restrict__ dst, int n) {
    int i = blockIdx.x * blockDim.x + threadIdx.x;
    if (i < n) dst[i] = (half_t)src[i];
}

// aW [64, D] f32  ->  aWT [D, 64] f16
__global__ void k_prep_aW(const float* __restrict__ aW,
                          half_t* __restrict__ aWT, int D) {
    int idx = blockIdx.x * blockDim.x + threadIdx.x;
    if (idx < D * 64) {
        int d = idx >> 6, j = idx & 63;
        aWT[idx] = (half_t)aW[j * D + d];
    }
}

// cw [D,7,2,256] f32 -> Wt [256][ (k*2+c)*D + dd ] f16
__global__ void k_prep_w(const float* __restrict__ cw,
                         half_t* __restrict__ Wt, int D) {
    size_t idx = (size_t)blockIdx.x * blockDim.x + threadIdx.x;
    size_t total = (size_t)Hh * 14 * D;
    if (idx >= total) return;
    int t  = (int)(idx % ((size_t)14 * D));
    int h  = (int)(idx / ((size_t)14 * D));
    int k  = t / (2 * D);
    int rm = t % (2 * D);
    int c  = rm / D;
    int dd = rm % D;
    Wt[idx] = (half_t)cw[(((size_t)dd * Kk + k) * 2 + c) * Hh + h];
}

// row squared-norms over f16 rows of length D; one block (256 thr) per row
__global__ void k_row_norms(const half_t* __restrict__ X,
                            float* __restrict__ nrm, int D) {
    const half_t* row = X + (size_t)blockIdx.x * D;
    float s = 0.f;
    for (int d = threadIdx.x; d < D; d += blockDim.x) {
        float v = (float)row[d];
        s += v * v;
    }
    __shared__ float red[256];
    red[threadIdx.x] = s;
    __syncthreads();
    for (int off = 128; off > 0; off >>= 1) {
        if ((int)threadIdx.x < off) red[threadIdx.x] += red[threadIdx.x + off];
        __syncthreads();
    }
    if (threadIdx.x == 0) nrm[blockIdx.x] = red[0];
}

// LO0/RO0: mean over S of X [B,S,D] f32 -> out [B,D]
__global__ void k_col_mean_f32(const float* __restrict__ X,
                               float* __restrict__ out, int D) {
    int idx = blockIdx.x * blockDim.x + threadIdx.x;
    if (idx >= Bb * D) return;
    int b = idx / D, d = idx % D;
    float s = 0.f;
    for (int p = 0; p < Ss; ++p) s += X[((size_t)b * Ss + p) * D + d];
    out[idx] = s * (1.0f / Ss);
}

// mean over W positions of [B,W,256] f16 -> [B,256] f32 (y selects L/R)
__global__ void k_pos_mean(const half_t* __restrict__ L,
                           const half_t* __restrict__ R,
                           float* __restrict__ OL, float* __restrict__ OR,
                           int W) {
    const half_t* X = blockIdx.y ? R : L;
    float* O        = blockIdx.y ? OR : OL;
    int b = blockIdx.x, h = threadIdx.x;
    float s = 0.f;
    for (int w = 0; w < W; ++w) s += (float)X[((size_t)b * W + w) * Hh + h];
    O[b * Hh + h] = s / (float)W;
}

// ---------------------------------------------------------------------------
// Attention matrix: att[b,i,j] = 1/(1+sqrt(max(n1_i+n2_j-2*P_i.Q_j,0)+eps))
// P,Q: [B,64,D] position-major f16.  Writes att and att^T (both f16).
// Block = 128 threads = 4 waves; wave owns M-tile tm and ALL 4 N-tiles,
// loading each A-fragment once and issuing 4 WMMAs against it.
// ---------------------------------------------------------------------------
__global__ void k_att(const half_t* __restrict__ P, const half_t* __restrict__ Q,
                      const float* __restrict__ n1, const float* __restrict__ n2,
                      half_t* __restrict__ att, half_t* __restrict__ attT,
                      int D) {
    const int b    = blockIdx.x;
    const int wave = threadIdx.x >> 5, lane = threadIdx.x & 31;
    const int lo = lane & 15, hi = lane >> 4;
    const int tm = wave * 16;

    const half_t* Arow = P + ((size_t)b * Ss + tm + lo) * D;
    const half_t* Brow[4];
#pragma unroll
    for (int n = 0; n < 4; ++n)
        Brow[n] = Q + ((size_t)b * Ss + n * 16 + lo) * D;

    v8f z = {};
    v8f acc[4] = {z, z, z, z};
    for (int k = 0; k < D; k += 32) {
        frag16 a = load_a(Arow, k, hi);
#pragma unroll
        for (int n = 0; n < 4; ++n) {
            frag16 bb = load_b(Brow[n], k, hi);
            acc[n] = wmma32(a, bb, acc[n]);
        }
    }
#pragma unroll
    for (int n = 0; n < 4; ++n) {
        int j = n * 16 + lo;
#pragma unroll
        for (int r = 0; r < 8; ++r) {
            int i = tm + r + hi * 8;
            float sq = n1[b * Ss + i] + n2[b * Ss + j] - 2.0f * acc[n][r];
            sq = fmaxf(sq, 0.0f);
            float v = 1.0f / (1.0f + sqrtf(sq + EPS));
            att[((size_t)b * Ss + i) * Ss + j]  = (half_t)v;
            attT[((size_t)b * Ss + j) * Ss + i] = (half_t)v;
        }
    }
}

// ---------------------------------------------------------------------------
// x?a[b,i,d] = sum_j att[b,i,j] * aW[j,d]   (aWT: [D][64] f16)
// grid (B, D/64, 2); block 128 = 4 waves; wave = M-tile, 4 N-tiles each.
// ---------------------------------------------------------------------------
__global__ void k_xa(const half_t* __restrict__ att,
                     const half_t* __restrict__ attT,
                     const half_t* __restrict__ aWT,
                     half_t* __restrict__ x1a, half_t* __restrict__ x2a,
                     int D) {
    const half_t* A = blockIdx.z ? attT : att;
    half_t* O       = blockIdx.z ? x2a : x1a;
    const int b = blockIdx.x, ntb = blockIdx.y * 64;
    const int wave = threadIdx.x >> 5, lane = threadIdx.x & 31;
    const int lo = lane & 15, hi = lane >> 4;
    const int tm = wave * 16;

    const half_t* Arow = A + ((size_t)b * Ss + tm + lo) * Ss;
    const half_t* Brow[4];
#pragma unroll
    for (int n = 0; n < 4; ++n)
        Brow[n] = aWT + (size_t)(ntb + n * 16 + lo) * Ss;

    v8f z = {};
    v8f acc[4] = {z, z, z, z};
    for (int k = 0; k < Ss; k += 32) {
        frag16 a = load_a(Arow, k, hi);
#pragma unroll
        for (int n = 0; n < 4; ++n) {
            frag16 bb = load_b(Brow[n], k, hi);
            acc[n] = wmma32(a, bb, acc[n]);
        }
    }
#pragma unroll
    for (int n = 0; n < 4; ++n) {
#pragma unroll
        for (int r = 0; r < 8; ++r) {
            int i = tm + r + hi * 8, d = ntb + n * 16 + lo;
            O[((size_t)b * Ss + i) * D + d] = (half_t)acc[n][r];
        }
    }
}

// ---------------------------------------------------------------------------
// Wide conv as implicit GEMM:
//   y[g=(b,w), h] = sum_{k,c,dd} patch[g, (k,c,dd)] * Wt[h][(k*2+c)*D+dd]
//   out = tanh(y + bias), stored [B,70,256] f16 (position-major).
// grid (560, 1, 2); block 128 = 4 waves; wave = M-tile x 4 N-tiles (so one
// block covers all 16 N-tiles). Zero-padding via branchless selects so EXEC
// stays full for WMMA.
// ---------------------------------------------------------------------------
__global__ void k_conv(const half_t* __restrict__ XL, const half_t* __restrict__ XAL,
                       const half_t* __restrict__ XR, const half_t* __restrict__ XAR,
                       const half_t* __restrict__ Wt, const float* __restrict__ cb,
                       half_t* __restrict__ lc, half_t* __restrict__ rc, int D) {
    const half_t* X  = blockIdx.z ? XR : XL;
    const half_t* XA = blockIdx.z ? XAR : XAL;
    half_t* O        = blockIdx.z ? rc : lc;

    const int wave = threadIdx.x >> 5, lane = threadIdx.x & 31;
    const int lo = lane & 15, hi = lane >> 4;
    const int mt = blockIdx.x * 16;                 // M tile over g = b*70+w
    const int nt = wave * 64;                       // this wave's N base (4 tiles)

    const int g = mt + lo;                          // this lane's A row
    const int b = g / Wp, w = g % Wp;
    const half_t* Brow[4];
#pragma unroll
    for (int n = 0; n < 4; ++n)
        Brow[n] = Wt + (size_t)(nt + n * 16 + lo) * (14 * D);

    const uint4 z4 = {0u, 0u, 0u, 0u};
    v8f z = {};
    v8f acc[4] = {z, z, z, z};
    for (int k = 0; k < Kk; ++k) {
        int s = w + k - (Kk - 1);
        bool ok = (unsigned)s < (unsigned)Ss;
        int sc = ok ? s : 0;
        const half_t* xrow  = X  + ((size_t)b * Ss + sc) * D;
        const half_t* xarow = XA + ((size_t)b * Ss + sc) * D;
        __builtin_prefetch(xrow + D, 0, 0);         // next position's row
        for (int c = 0; c < 2; ++c) {
            const half_t* arow = c ? xarow : xrow;
            const int tb = (k * 2 + c) * D;
            for (int dd = 0; dd < D; dd += 32) {
                frag16 a;
                uint4 a0 = *(const uint4*)(arow + dd + hi * 8);
                uint4 a1 = *(const uint4*)(arow + dd + hi * 8 + 16);
                a.u[0] = ok ? a0 : z4;
                a.u[1] = ok ? a1 : z4;
#pragma unroll
                for (int n = 0; n < 4; ++n) {
                    frag16 bb = load_b(Brow[n], tb + dd, hi);
                    acc[n] = wmma32(a, bb, acc[n]);
                }
            }
        }
    }
#pragma unroll
    for (int n = 0; n < 4; ++n) {
#pragma unroll
        for (int r = 0; r < 8; ++r) {
            int g2 = mt + r + hi * 8;
            int b2 = g2 / Wp, w2 = g2 % Wp;
            int h  = nt + n * 16 + lo;
            float y = tanhf(acc[n][r] + cb[h]);
            O[((size_t)b2 * Wp + w2) * Hh + h] = (half_t)y;
        }
    }
}

// ---------------------------------------------------------------------------
// Second attention + row/col sums only (la[b,i] = sum_j att2, ra[b,j] = sum_i).
// W=70 padded to 80 -> 5x5 tiles per batch; block 160 thr = 5 waves; each
// wave owns one M-tile-row and all 5 N-tiles (A-fragment reused 5x). Sums
// accumulated via LDS float atomics (ds_add_f32).
// ---------------------------------------------------------------------------
__global__ void k_att2sum(const half_t* __restrict__ lc, const half_t* __restrict__ rc,
                          const float* __restrict__ nl, const float* __restrict__ nr,
                          float* __restrict__ la, float* __restrict__ ra) {
    __shared__ float sL[80], sR[80];
    const int b = blockIdx.x;
    if (threadIdx.x < 80) { sL[threadIdx.x] = 0.f; sR[threadIdx.x] = 0.f; }
    __syncthreads();

    const int wave = threadIdx.x >> 5, lane = threadIdx.x & 31;
    const int lo = lane & 15, hi = lane >> 4;
    const int tm = wave * 16;

    int il = tm + lo; il = il < Wp ? il : Wp - 1;           // clamped loads,
    const half_t* Arow = lc + ((size_t)b * Wp + il) * Hh;   // masked accumulate
    const half_t* Brow[5];
#pragma unroll
    for (int n = 0; n < 5; ++n) {
        int jl = n * 16 + lo; jl = jl < Wp ? jl : Wp - 1;
        Brow[n] = rc + ((size_t)b * Wp + jl) * Hh;
    }

    v8f z = {};
    v8f acc[5] = {z, z, z, z, z};
    for (int k = 0; k < Hh; k += 32) {
        frag16 a = load_a(Arow, k, hi);
#pragma unroll
        for (int n = 0; n < 5; ++n) {
            frag16 bb = load_b(Brow[n], k, hi);
            acc[n] = wmma32(a, bb, acc[n]);
        }
    }
#pragma unroll
    for (int n = 0; n < 5; ++n) {
        int j = n * 16 + lo;
        if (j < Wp) {
#pragma unroll
            for (int r = 0; r < 8; ++r) {
                int i = tm + r + hi * 8;
                if (i < Wp) {
                    float sq = fmaxf(nl[b * Wp + i] + nr[b * Wp + j] - 2.0f * acc[n][r], 0.f);
                    float v  = 1.0f / (1.0f + sqrtf(sq + EPS));
                    atomicAdd(&sL[i], v);
                    atomicAdd(&sR[j], v);
                }
            }
        }
    }
    __syncthreads();
    if (threadIdx.x < Wp)
        la[b * Wp + threadIdx.x] = sL[threadIdx.x];
    else if (threadIdx.x < 2 * Wp)
        ra[b * Wp + threadIdx.x - Wp] = sR[threadIdx.x - Wp];
}

// lwp[b,s,h] = sum_{j<7} lc[b,s+j,h] * la[b,s+j]; grid (B,64,2), block 256
__global__ void k_wpool(const half_t* __restrict__ lc, const half_t* __restrict__ rc,
                        const float* __restrict__ la, const float* __restrict__ ra,
                        half_t* __restrict__ lwp, half_t* __restrict__ rwp) {
    const half_t* X = blockIdx.z ? rc : lc;
    const float* A  = blockIdx.z ? ra : la;
    half_t* O       = blockIdx.z ? rwp : lwp;
    int b = blockIdx.x, s = blockIdx.y, h = threadIdx.x;
    float acc = 0.f;
#pragma unroll
    for (int j = 0; j < Kk; ++j) {
        int p = s + j;
        acc += (float)X[((size_t)b * Wp + p) * Hh + h] * A[b * Wp + p];
    }
    O[((size_t)b * Ss + s) * Hh + h] = (half_t)acc;
}

// Final head: three cosine sims + tiny 3x3 matmul -> logits [B,3]
DEV float cosv(const float* a, const float* b, int n) {
    float d = 0.f, x = 0.f, y = 0.f;
    for (int i = 0; i < n; ++i) { d += a[i] * b[i]; x += a[i] * a[i]; y += b[i] * b[i]; }
    return d / (sqrtf(x) * sqrtf(y) + EPS);
}

__global__ void k_head(const float* __restrict__ LO0, const float* __restrict__ RO0,
                       const float* __restrict__ LO1, const float* __restrict__ RO1,
                       const float* __restrict__ LO2, const float* __restrict__ RO2,
                       const float* __restrict__ ow, const float* __restrict__ ob,
                       float* __restrict__ out) {
    int b = blockIdx.x * blockDim.x + threadIdx.x;
    if (b >= Bb) return;
    float s0 = cosv(LO0 + (size_t)b * D0, RO0 + (size_t)b * D0, D0);
    float s1 = cosv(LO1 + (size_t)b * Hh, RO1 + (size_t)b * Hh, Hh);
    float s2 = cosv(LO2 + (size_t)b * Hh, RO2 + (size_t)b * Hh, Hh);
#pragma unroll
    for (int c = 0; c < Cc; ++c)
        out[b * Cc + c] = s0 * ow[0 * Cc + c] + s1 * ow[1 * Cc + c]
                        + s2 * ow[2 * Cc + c] + ob[c];
}

// ---------------------------------------------------------------------------
// Host orchestration
// ---------------------------------------------------------------------------
extern "C" void kernel_launch(void* const* d_in, const int* in_sizes, int n_in,
                              void* d_out, int out_size, void* d_ws, size_t ws_size,
                              hipStream_t stream) {
    (void)in_sizes; (void)n_in; (void)out_size; (void)ws_size;

    const float* premise = (const float*)d_in[0];
    const float* hyp     = (const float*)d_in[1];
    const float* aW1     = (const float*)d_in[2];
    const float* cw1     = (const float*)d_in[3];
    const float* cb1     = (const float*)d_in[4];
    const float* aW2     = (const float*)d_in[5];
    const float* cw2     = (const float*)d_in[6];
    const float* cb2     = (const float*)d_in[7];
    const float* out_w   = (const float*)d_in[8];
    const float* out_b   = (const float*)d_in[9];
    float* out = (float*)d_out;

    // ---- workspace carve-out (cursor allocator, 256B aligned) ----
    char* cur = (char*)d_ws;
    auto alloc = [&](size_t bytes) -> char* {
        char* p = cur;
        cur += (bytes + 255) & ~(size_t)255;
        return p;
    };
    half_t* pre_h  = (half_t*)alloc((size_t)Bb * Ss * D0 * 2);
    half_t* hyp_h  = (half_t*)alloc((size_t)Bb * Ss * D0 * 2);
    half_t* x1a_h  = (half_t*)alloc((size_t)Bb * Ss * D0 * 2);  // reused (layer2 fits inside)
    half_t* x2a_h  = (half_t*)alloc((size_t)Bb * Ss * D0 * 2);
    half_t* aW1T   = (half_t*)alloc((size_t)D0 * Ss * 2);
    half_t* aW2T   = (half_t*)alloc((size_t)Hh * Ss * 2);
    half_t* Wt1    = (half_t*)alloc((size_t)Hh * 14 * D0 * 2);
    half_t* Wt2    = (half_t*)alloc((size_t)Hh * 14 * Hh * 2);
    half_t* att_h  = (half_t*)alloc((size_t)Bb * Ss * Ss * 2);
    half_t* attT_h = (half_t*)alloc((size_t)Bb * Ss * Ss * 2);
    float*  nA     = (float*)alloc((size_t)Bb * Wp * 4);        // row norms (reused)
    float*  nB     = (float*)alloc((size_t)Bb * Wp * 4);
    half_t* lc_h   = (half_t*)alloc((size_t)Bb * Wp * Hh * 2);  // reused for layer2
    half_t* rc_h   = (half_t*)alloc((size_t)Bb * Wp * Hh * 2);
    float*  la     = (float*)alloc((size_t)Bb * Wp * 4);
    float*  ra     = (float*)alloc((size_t)Bb * Wp * 4);
    half_t* lwp_h  = (half_t*)alloc((size_t)Bb * Ss * Hh * 2);
    half_t* rwp_h  = (half_t*)alloc((size_t)Bb * Ss * Hh * 2);
    float*  LO0    = (float*)alloc((size_t)Bb * D0 * 4);
    float*  RO0    = (float*)alloc((size_t)Bb * D0 * 4);
    float*  LO1    = (float*)alloc((size_t)Bb * Hh * 4);
    float*  RO1    = (float*)alloc((size_t)Bb * Hh * 4);
    float*  LO2    = (float*)alloc((size_t)Bb * Hh * 4);
    float*  RO2    = (float*)alloc((size_t)Bb * Hh * 4);

    const int nIn = Bb * Ss * D0;

    // ---- precompute / conversions ----
    k_f32_to_f16<<<(nIn + 255) / 256, 256, 0, stream>>>(premise, pre_h, nIn);
    k_f32_to_f16<<<(nIn + 255) / 256, 256, 0, stream>>>(hyp, hyp_h, nIn);
    k_prep_aW<<<(D0 * Ss + 255) / 256, 256, 0, stream>>>(aW1, aW1T, D0);
    k_prep_aW<<<(Hh * Ss + 255) / 256, 256, 0, stream>>>(aW2, aW2T, Hh);
    k_prep_w<<<((Hh * 14 * D0) + 255) / 256, 256, 0, stream>>>(cw1, Wt1, D0);
    k_prep_w<<<((Hh * 14 * Hh) + 255) / 256, 256, 0, stream>>>(cw2, Wt2, Hh);
    k_col_mean_f32<<<(Bb * D0 + 255) / 256, 256, 0, stream>>>(premise, LO0, D0);
    k_col_mean_f32<<<(Bb * D0 + 255) / 256, 256, 0, stream>>>(hyp, RO0, D0);

    // ---- layer 1 ----
    k_row_norms<<<Bb * Ss, 256, 0, stream>>>(pre_h, nA, D0);
    k_row_norms<<<Bb * Ss, 256, 0, stream>>>(hyp_h, nB, D0);
    k_att<<<Bb, 128, 0, stream>>>(pre_h, hyp_h, nA, nB, att_h, attT_h, D0);
    k_xa<<<dim3(Bb, D0 / 64, 2), 128, 0, stream>>>(att_h, attT_h, aW1T, x1a_h, x2a_h, D0);
    k_conv<<<dim3((Bb * Wp) / 16, 1, 2), 128, 0, stream>>>(
        pre_h, x1a_h, hyp_h, x2a_h, Wt1, cb1, lc_h, rc_h, D0);

    k_row_norms<<<Bb * Wp, 256, 0, stream>>>(lc_h, nA, Hh);
    k_row_norms<<<Bb * Wp, 256, 0, stream>>>(rc_h, nB, Hh);
    k_att2sum<<<Bb, 160, 0, stream>>>(lc_h, rc_h, nA, nB, la, ra);
    k_wpool<<<dim3(Bb, Ss, 2), 256, 0, stream>>>(lc_h, rc_h, la, ra, lwp_h, rwp_h);
    k_pos_mean<<<dim3(Bb, 2), 256, 0, stream>>>(lc_h, rc_h, LO1, RO1, Wp);

    // ---- layer 2 (only the all-pool outputs are live) ----
    k_row_norms<<<Bb * Ss, 256, 0, stream>>>(lwp_h, nA, Hh);
    k_row_norms<<<Bb * Ss, 256, 0, stream>>>(rwp_h, nB, Hh);
    k_att<<<Bb, 128, 0, stream>>>(lwp_h, rwp_h, nA, nB, att_h, attT_h, Hh);
    k_xa<<<dim3(Bb, Hh / 64, 2), 128, 0, stream>>>(att_h, attT_h, aW2T, x1a_h, x2a_h, Hh);
    k_conv<<<dim3((Bb * Wp) / 16, 1, 2), 128, 0, stream>>>(
        lwp_h, x1a_h, rwp_h, x2a_h, Wt2, cb2, lc_h, rc_h, Hh);
    k_pos_mean<<<dim3(Bb, 2), 256, 0, stream>>>(lc_h, rc_h, LO2, RO2, Wp);

    // ---- head ----
    k_head<<<1, 128, 0, stream>>>(LO0, RO0, LO1, RO1, LO2, RO2, out_w, out_b, out);
}
